// ODEBlockwScore_41635412967614
// MI455X (gfx1250) — matrix operand order repaired and checked by reading, not compiled, not run on hardware
//
#include <hip/hip_runtime.h>
#include <hip/hip_bf16.h>
#include <math.h>

// ODE CNF-with-score block, MI455X (gfx1250, wave32, WMMA).
// Fixed-step RK4 (32 steps) replaces adaptive dopri5 (rtol=atol=1e-3).
// All 128x128 matmuls run as v_wmma_f32_16x16x32_f16 with f32 accumulation.
// Weights are pre-swizzled in LDS into B-fragment-native layout (one
// contiguous 32B run per lane per k-chunk/column-tile) for both W and W^T,
// so every WMMA operand load is a pair of ds_load_b128.

#define DD      3
#define FEAT    128
#define TB      16      // batch elements per workgroup (one WMMA M-tile)
#define PIT     136     // LDS row pitch in halves for activations (padding)
#define NSTEPS  32
#define NTH     256
#define BATCH   8192
#define WFRAG   (4*8*32*16)   // 4 k-chunks x 8 col-tiles x 32 lanes x 16 halves

typedef __attribute__((ext_vector_type(16))) _Float16 v16h;
typedef __attribute__((ext_vector_type(8)))  float    v8f;

struct Smem {
  float sb0[FEAT], sb1[FEAT], sb2[FEAT], sb3[4];
  float sstate[TB*8], sstage[TB*8], sksum[TB*8], souts[TB*8];
  float strace[TB];
  _Float16 sW0[4*FEAT];       // W0 rows: [t, x0, x1, x2] x 128
  _Float16 sW3T[DD*FEAT];     // W3 columns: sW3T[j*FEAT+k] = W3[k][j]
  _Float16 sW1B[WFRAG];       // W1  in B-fragment layout
  _Float16 sW2B[WFRAG];       // W2  in B-fragment layout
  _Float16 sW1TB[WFRAG];      // W1^T in B-fragment layout
  _Float16 sW2TB[WFRAG];      // W2^T in B-fragment layout
  _Float16 sAct1[(4*TB)*PIT]; // rows 0..15: h1 ; rows 16+j*16+i: u1^(j) elem i
  _Float16 sAct2[(4*TB)*PIT]; // same layout for layer 2
  _Float16 bufA[(5*TB)*PIT];  // backward GEMM A: [q | rho0 | rho1 | rho2 | czz]
  _Float16 bufO[(5*TB)*PIT];  // GEMM outputs (f16 staging)
};

// O(rows x 128) = A(rows x 128) * B(128 x 128) using v_wmma_f32_16x16x32_f16.
// A/O rows in LDS at pitch PIT; WB is a pre-swizzled B-fragment array.
__device__ __forceinline__ void gemm_lds(const _Float16* __restrict__ A,
                                         const _Float16* __restrict__ WB,
                                         _Float16* __restrict__ O,
                                         int rowtiles, int wave, int lane) {
  const int J  = rowtiles * 8;       // 8 column tiles of 16
  const int m  = lane & 15;
  const int hi = lane >> 4;
  for (int job = wave; job < J; job += 8) {
    const int rt = job >> 3;
    const int ct = job & 7;
    v8f acc = {};
    #pragma unroll
    for (int kk = 0; kk < 4; ++kk) {
      v16h a, b;
      // A frag: lane m holds K = {0..15} (hi=0) or {16..31} (hi=1)
      const _Float16* pa = A + (rt*16 + m)*PIT + kk*32 + hi*16;
      #pragma unroll
      for (int e = 0; e < 8; ++e) { a[e] = pa[e]; a[8+e] = pa[8+e]; }
      // B frag: contiguous 32B per lane (pre-swizzled)
      const _Float16* pb = WB + ((kk*8 + ct)*32 + lane)*16;
      #pragma unroll
      for (int e = 0; e < 16; ++e) b[e] = pb[e];
      acc = __builtin_amdgcn_wmma_f32_16x16x32_f16(false, a, false, b,
                                                   (short)0, acc, false, false);
    }
    // D frag: lane col n = ct*16+m, rows v + 8*hi
    #pragma unroll
    for (int v = 0; v < 8; ++v)
      O[(rt*16 + hi*8 + v)*PIT + ct*16 + m] = (_Float16)acc[v];
  }
}

__global__ __launch_bounds__(NTH, 1)
void ode_cnf_score_kernel(const float* __restrict__ states,
                          const float* __restrict__ W0, const float* __restrict__ b0,
                          const float* __restrict__ W1, const float* __restrict__ b1,
                          const float* __restrict__ W2, const float* __restrict__ b2,
                          const float* __restrict__ W3, const float* __restrict__ b3,
                          float* __restrict__ out) {
  extern __shared__ unsigned char smem_raw[];
  Smem& S = *reinterpret_cast<Smem*>(smem_raw);
  const int tid  = threadIdx.x;
  const int lane = tid & 31;
  const int wave = tid >> 5;
  const int e0   = blockIdx.x * TB;

  // ---- stage weights into LDS, swizzled to B-fragment layout ----
  for (int idx = tid; idx < WFRAG; idx += NTH) {
    const int e  = idx & 15;
    const int ln = (idx >> 4) & 31;
    const int ct = (idx >> 9) & 7;
    const int kk = idx >> 12;
    const int n  = ct*16 + (ln & 15);
    const int k  = kk*32 + ((ln >> 4) << 4) + e;
    S.sW1B[idx]  = (_Float16)W1[k*FEAT + n];   // B = W1
    S.sW2B[idx]  = (_Float16)W2[k*FEAT + n];   // B = W2
    S.sW1TB[idx] = (_Float16)W1[n*FEAT + k];   // B = W1^T
    S.sW2TB[idx] = (_Float16)W2[n*FEAT + k];   // B = W2^T
  }
  for (int idx = tid; idx < 4*FEAT; idx += NTH) S.sW0[idx] = (_Float16)W0[idx];
  for (int idx = tid; idx < DD*FEAT; idx += NTH) {
    const int j = idx >> 7, k = idx & 127;
    S.sW3T[idx] = (_Float16)W3[k*DD + j];
  }
  for (int n = tid; n < FEAT; n += NTH) { S.sb0[n]=b0[n]; S.sb1[n]=b1[n]; S.sb2[n]=b2[n]; }
  if (tid < 4) S.sb3[tid] = (tid < DD) ? b3[tid] : 0.f;
  for (int idx = tid; idx < TB*8; idx += NTH) {
    const int i = idx >> 3, c = idx & 7;
    S.sstate[idx] = (c < 7) ? states[(size_t)(e0+i)*7 + c] : 0.f;
  }
  __syncthreads();

  const float T0c   = 1.0f;
  const float hstep = 1.0f / (float)NSTEPS;

  for (int step = 0; step < NSTEPS; ++step) {
    for (int idx = tid; idx < TB*8; idx += NTH) { S.sstage[idx]=S.sstate[idx]; S.sksum[idx]=0.f; }
    __syncthreads();

    for (int stg = 0; stg < 4; ++stg) {
      const float toff   = (stg == 0) ? 0.f : (stg == 3 ? 1.f : 0.5f);
      const float t_eval = T0c * ((float)step + toff) * hstep;

      // ======== EVAL: souts = f_ode(sstage, t_eval) ========
      for (int idx = tid; idx < TB*8; idx += NTH) S.souts[idx] = 0.f;
      if (tid < TB) S.strace[tid] = 0.f;
      __syncthreads();

      // layer 1 forward (K=4, VALU) + tangent seeds u1^(j) = W0[1+j,:] * d1
      for (int idx = tid; idx < TB*FEAT; idx += NTH) {
        const int i = idx >> 7, n = idx & 127;
        const float x0 = S.sstage[i*8+0], x1 = S.sstage[i*8+1], x2 = S.sstage[i*8+2];
        const float z = S.sb0[n] + t_eval*(float)S.sW0[n]
                      + x0*(float)S.sW0[FEAT+n] + x1*(float)S.sW0[2*FEAT+n]
                      + x2*(float)S.sW0[3*FEAT+n];
        const float h1 = tanhf(z);
        const float d1 = 1.f - h1*h1;
        S.sAct1[i*PIT + n] = (_Float16)h1;
        S.sAct1[(16 + 0*16 + i)*PIT + n] = (_Float16)((float)S.sW0[FEAT+n]   * d1);
        S.sAct1[(16 + 1*16 + i)*PIT + n] = (_Float16)((float)S.sW0[2*FEAT+n] * d1);
        S.sAct1[(16 + 2*16 + i)*PIT + n] = (_Float16)((float)S.sW0[3*FEAT+n] * d1);
      }
      __syncthreads();
      gemm_lds(S.sAct1, S.sW1B, S.bufO, 4, wave, lane);   // [h1;u1] * W1
      __syncthreads();

      // layer 2 activation: h2 = tanh(pre+b1); u2 = pre_u * d2
      for (int idx = tid; idx < TB*FEAT; idx += NTH) {
        const int i = idx >> 7, n = idx & 127;
        const float h2 = tanhf((float)S.bufO[i*PIT+n] + S.sb1[n]);
        const float d2 = 1.f - h2*h2;
        S.sAct2[i*PIT + n] = (_Float16)h2;
        #pragma unroll
        for (int j = 0; j < DD; ++j)
          S.sAct2[(16 + j*16 + i)*PIT + n] =
              (_Float16)((float)S.bufO[(16 + j*16 + i)*PIT + n] * d2);
      }
      __syncthreads();
      gemm_lds(S.sAct2, S.sW2B, S.bufO, 4, wave, lane);   // [h2;u2] * W2
      __syncthreads();

      // layer 3: head out = h3*W3+b3, trace = sum_i <u3^(i), W3[:,i]>,
      // backward seeds: q3 = (W3*score)*d3, rho3^(i) = W3[:,i]*d3,
      // cz3 = -2*h3 * sum_i W3[:,i]*u3^(i)
      for (int idx = tid; idx < TB*FEAT; idx += NTH) {
        const int i = idx >> 7, n = idx & 127;
        const float h3 = tanhf((float)S.bufO[i*PIT+n] + S.sb2[n]);
        const float d3 = 1.f - h3*h3;
        const float w0t = (float)S.sW3T[n];
        const float w1t = (float)S.sW3T[FEAT+n];
        const float w2t = (float)S.sW3T[2*FEAT+n];
        const float u30 = (float)S.bufO[(16+0*16+i)*PIT+n] * d3;
        const float u31 = (float)S.bufO[(16+1*16+i)*PIT+n] * d3;
        const float u32 = (float)S.bufO[(16+2*16+i)*PIT+n] * d3;
        const float sc0 = S.sstage[i*8+4], sc1 = S.sstage[i*8+5], sc2 = S.sstage[i*8+6];
        S.bufA[i*PIT + n]             = (_Float16)((sc0*w0t + sc1*w1t + sc2*w2t) * d3);
        S.bufA[(16+0*16+i)*PIT + n]   = (_Float16)(w0t * d3);
        S.bufA[(16+1*16+i)*PIT + n]   = (_Float16)(w1t * d3);
        S.bufA[(16+2*16+i)*PIT + n]   = (_Float16)(w2t * d3);
        S.bufA[(64+i)*PIT + n]        = (_Float16)(-2.f*h3*(w0t*u30 + w1t*u31 + w2t*u32));
        atomicAdd(&S.strace[i], u30*w0t + u31*w1t + u32*w2t);
        atomicAdd(&S.souts[i*8+0], h3*w0t);
        atomicAdd(&S.souts[i*8+1], h3*w1t);
        atomicAdd(&S.souts[i*8+2], h3*w2t);
      }
      __syncthreads();
      gemm_lds(S.bufA, S.sW2TB, S.bufO, 5, wave, lane);   // [q|rho|czz] * W2^T
      __syncthreads();

      // layer 2 backward: q2 = qt*d2; rho2 = r2*d2; czz2 = ct*d2 - 2h2*sum r2*u2
      for (int idx = tid; idx < TB*FEAT; idx += NTH) {
        const int i = idx >> 7, n = idx & 127;
        const float h2 = (float)S.sAct2[i*PIT+n];
        const float d2 = 1.f - h2*h2;
        float czv = 0.f;
        #pragma unroll
        for (int j = 0; j < DD; ++j) {
          const float r2 = (float)S.bufO[(16+j*16+i)*PIT+n];
          const float u2 = (float)S.sAct2[(16+j*16+i)*PIT+n];
          czv += r2*u2;
          S.bufA[(16+j*16+i)*PIT + n] = (_Float16)(r2 * d2);
        }
        S.bufA[i*PIT + n]      = (_Float16)((float)S.bufO[i*PIT+n] * d2);
        S.bufA[(64+i)*PIT + n] = (_Float16)((float)S.bufO[(64+i)*PIT+n] * d2 - 2.f*h2*czv);
      }
      __syncthreads();
      gemm_lds(S.bufA, S.sW1TB, S.bufO, 5, wave, lane);   // ... * W1^T
      __syncthreads();

      // layer 1 backward: dscore_j = sum_n (czz1 - q1) * W0[1+j][n]
      for (int idx = tid; idx < TB*FEAT; idx += NTH) {
        const int i = idx >> 7, n = idx & 127;
        const float h1 = (float)S.sAct1[i*PIT+n];
        const float d1 = 1.f - h1*h1;
        float czv = 0.f;
        #pragma unroll
        for (int j = 0; j < DD; ++j) {
          const float r1 = (float)S.bufO[(16+j*16+i)*PIT+n];
          const float u1 = (float)S.sAct1[(16+j*16+i)*PIT+n];
          czv += r1*u1;
        }
        const float q1   = (float)S.bufO[i*PIT+n] * d1;
        const float czz1 = (float)S.bufO[(64+i)*PIT+n] * d1 - 2.f*h1*czv;
        const float coeff = czz1 - q1;
        atomicAdd(&S.souts[i*8+4], coeff*(float)S.sW0[FEAT+n]);
        atomicAdd(&S.souts[i*8+5], coeff*(float)S.sW0[2*FEAT+n]);
        atomicAdd(&S.souts[i*8+6], coeff*(float)S.sW0[3*FEAT+n]);
      }
      __syncthreads();
      if (tid < TB) {
        S.souts[tid*8+0] += S.sb3[0];
        S.souts[tid*8+1] += S.sb3[1];
        S.souts[tid*8+2] += S.sb3[2];
        S.souts[tid*8+3]  = -S.strace[tid];
      }
      __syncthreads();

      // ======== RK4 combine ========
      const float wgt   = (stg == 0 || stg == 3) ? 1.f : 2.f;
      const float anext = (stg == 2) ? hstep : 0.5f*hstep;
      for (int idx = tid; idx < TB*8; idx += NTH) {
        if ((idx & 7) < 7) {
          const float k = T0c * S.souts[idx];
          S.sksum[idx] += wgt * k;
          if (stg < 3) S.sstage[idx] = S.sstate[idx] + anext * k;
        }
      }
      __syncthreads();
    }
    for (int idx = tid; idx < TB*8; idx += NTH)
      if ((idx & 7) < 7) S.sstate[idx] += (hstep/6.f) * S.sksum[idx];
    __syncthreads();
  }

  for (int idx = tid; idx < TB*7; idx += NTH) {
    const int i = idx / 7, c = idx % 7;
    out[(size_t)(e0+i)*7 + c] = S.sstate[i*8 + c];
  }
}

extern "C" void kernel_launch(void* const* d_in, const int* in_sizes, int n_in,
                              void* d_out, int out_size, void* d_ws, size_t ws_size,
                              hipStream_t stream) {
  (void)in_sizes; (void)n_in; (void)out_size; (void)d_ws; (void)ws_size;
  const float* states = (const float*)d_in[0];
  const float* W0 = (const float*)d_in[1];
  const float* b0 = (const float*)d_in[2];
  const float* W1 = (const float*)d_in[3];
  const float* b1 = (const float*)d_in[4];
  const float* W2 = (const float*)d_in[5];
  const float* b2 = (const float*)d_in[6];
  const float* W3 = (const float*)d_in[7];
  const float* b3 = (const float*)d_in[8];
  float* out = (float*)d_out;

  const size_t shmem = sizeof(Smem);  // ~210 KB; ISA allows up to 320 KB/WG
  hipFuncSetAttribute(reinterpret_cast<const void*>(ode_cnf_score_kernel),
                      hipFuncAttributeMaxDynamicSharedMemorySize, (int)shmem);
  ode_cnf_score_kernel<<<BATCH/TB, NTH, shmem, stream>>>(
      states, W0, b0, W1, b1, W2, b2, W3, b3, out);
}
